// InputDependentBend_61753039781992
// MI455X (gfx1250) — matrix operand order repaired and checked by reading, not compiled, not run on hardware
//
#include <hip/hip_runtime.h>
#include <hip/hip_bf16.h>
#include <math.h>

// ---------------------------------------------------------------------------
// Problem constants (reference: B=8, N=8192, D=256, W=3)
// ---------------------------------------------------------------------------
#define TOK    65536     // B*N tokens
#define DMODEL 256
#define SEQ    8192      // per-sequence length (power of two, tiles never cross)
#define DFFN   1024

typedef __bf16 bf16x16 __attribute__((ext_vector_type(16)));
typedef float  f32x8   __attribute__((ext_vector_type(8)));
typedef unsigned int u32;

__device__ __forceinline__ float geluf(float v) {
    return 0.5f * v * (1.0f + erff(v * 0.70710678118654752f));
}

// ---------------------------------------------------------------------------
// f32 -> bf16 convert (for activations x)
// ---------------------------------------------------------------------------
__global__ __launch_bounds__(256) void cvt_bf16_k(const float* __restrict__ in,
                                                  __bf16* __restrict__ out, int n) {
    int i = blockIdx.x * 256 + threadIdx.x;
    if (i < n) out[i] = (__bf16)in[i];
}

// ---------------------------------------------------------------------------
// Weight shuffle: f32 [K x NN] row-major  ->  bf16 WMMA B-fragment records.
// Record for (kt, nt) tile = 32 lanes x 32 bytes, contiguous (1 KB):
//   lane l, dword v (0..7): elements B[kt*32 + (l>>4)*16 + 2v + {0,1}][nt*16 + (l&15)]
// Record order: tile = kt*(NN/16) + nt.  Linear u32 id = (tile*32 + lane)*8 + v.
// ---------------------------------------------------------------------------
__global__ __launch_bounds__(256) void shuffle_b_k(const float* __restrict__ W,
                                                   __bf16* __restrict__ out,
                                                   int K, int NN) {
    const int id = blockIdx.x * 256 + threadIdx.x;
    const int total = K * NN / 2;
    if (id >= total) return;
    const int v    = id & 7;
    const int lane = (id >> 3) & 31;
    const int tile = id >> 8;
    const int nt   = tile % (NN / 16);
    const int kt   = tile / (NN / 16);
    const int n = nt * 16 + (lane & 15);
    const int k = kt * 32 + (lane >> 4) * 16 + 2 * v;
    out[(size_t)id * 2 + 0] = (__bf16)W[(size_t)k * NN + n];
    out[(size_t)id * 2 + 1] = (__bf16)W[(size_t)(k + 1) * NN + n];
}

// ---------------------------------------------------------------------------
// Fused bf16 WMMA GEMM, block tile 64(M) x 256(N), 8 waves, no main-loop LDS.
// Wave tile 32x64: 2 M-subtiles x 4 N-tiles, explicit double buffering.
//   EPI 0: out_bf = bf16(gelu(acc + bias))                       (h, ffn1)
//   EPI 1: out_bf = bf16(x * (acc + bias))                       (warp gate)
//   EPI 2: y = LN(x + acc + bias) -> outF (f32) + outBF (bf16)   (mix+LN1)
//   EPI 3: out = LN(yres + acc + bias) -> outF (f32)             (ffn2+LN2)
// SHIFTED: A rows come from xw shifted by (slice-3) within each sequence,
//          slice = kglobal/256; rows before the sequence start read as zero.
// ---------------------------------------------------------------------------
#define CS_ST 260   // Cs row stride (floats)

enum { EPI_GELU = 0, EPI_WARP = 1, EPI_MIX_LN = 2, EPI_FFN_LN = 3 };

template <int EPI, bool SHIFTED, int K, int NN>
__global__ __launch_bounds__(256) void gemm64x256_k(
    const __bf16* __restrict__ A,     // [M x K] bf16 row-major (SHIFTED: [M x 256])
    const __bf16* __restrict__ Bs,    // shuffled weights (see shuffle_b_k)
    const float*  __restrict__ bias,  // [NN]
    const float*  __restrict__ resid, // x (EPI 1/2) or y_f32 (EPI 3), [M x 256]
    const float*  __restrict__ gamma, // LN gain  (EPI 2/3)
    const float*  __restrict__ beta,  // LN shift (EPI 2/3)
    float*        __restrict__ outF,  // f32 out (EPI 2/3)
    __bf16*       __restrict__ outBF) // bf16 out (EPI 0/1/2)
{
    extern __shared__ char smem[];
    float* Cs  = (float*)smem;        // 64 x CS_ST   (LN epilogues only)
    float* red = Cs + 64 * CS_ST;     // 64*4*2

    const int tid  = threadIdx.x;
    const int lane = tid & 31;
    const int wave = tid >> 5;
    const int mg   = wave >> 2;        // 0..1 : 32-row M group
    const int ng   = wave & 3;         // 0..3 : 64-col N group
    const int l15  = lane & 15;
    const int hi   = lane >> 4;
    const int m0   = blockIdx.x * 64;
    const int n0   = blockIdx.y * 256;

    const int row0   = m0 + mg * 32 + l15;   // A row for M-subtile 0 (subtile 1 = +16)
    const int seqpos = row0 & (SEQ - 1);
    const int ntg0   = (n0 >> 4) + ng * 4;   // base n-tile of this wave's 4 fragments

    f32x8 acc[2][4] = {};

    auto loadA = [&](int kt, bf16x16* aF) {
#pragma unroll
        for (int ms = 0; ms < 2; ++ms) {
            uint4* a4 = (uint4*)&aF[ms];
            if constexpr (SHIFTED) {
                const int s    = (kt * 32) >> 8;     // window slice 0..3
                const int col0 = (kt * 32) & 255;    // column within slice
                if (seqpos + ms * 16 + s - 3 >= 0) {
                    const __bf16* ap = A +
                        (size_t)(row0 + ms * 16 + s - 3) * DMODEL + col0 + hi * 16;
                    a4[0] = *(const uint4*)ap;
                    a4[1] = *(const uint4*)(ap + 8);
                } else {
                    a4[0] = make_uint4(0u, 0u, 0u, 0u);
                    a4[1] = make_uint4(0u, 0u, 0u, 0u);
                }
            } else {
                const __bf16* ap = A + (size_t)(row0 + ms * 16) * K + kt * 32 + hi * 16;
                a4[0] = *(const uint4*)ap;
                a4[1] = *(const uint4*)(ap + 8);
            }
        }
    };
    auto loadB = [&](int kt, bf16x16* bF) {
#pragma unroll
        for (int nt = 0; nt < 4; ++nt) {
            const size_t tile = (size_t)kt * (NN / 16) + ntg0 + nt;
            const __bf16* bp  = Bs + tile * 512 + lane * 16;
            uint4* b4 = (uint4*)&bF[nt];
            b4[0] = *(const uint4*)bp;
            b4[1] = *(const uint4*)(bp + 8);
        }
    };

    constexpr int KT = K / 32;
    bf16x16 aF[2][2], bF[2][4];
    loadA(0, aF[0]);
    loadB(0, bF[0]);
#pragma unroll 8
    for (int kt = 0; kt < KT - 1; ++kt) {
        const int cb = kt & 1, nb = cb ^ 1;
        loadA(kt + 1, aF[nb]);
        loadB(kt + 1, bF[nb]);
#pragma unroll
        for (int ms = 0; ms < 2; ++ms)
#pragma unroll
            for (int nt = 0; nt < 4; ++nt)
                acc[ms][nt] = __builtin_amdgcn_wmma_f32_16x16x32_bf16(
                    false, aF[cb][ms], false, bF[cb][nt], (short)0,
                    acc[ms][nt], false, false);
    }
    {
        const int cb = (KT - 1) & 1;
#pragma unroll
        for (int ms = 0; ms < 2; ++ms)
#pragma unroll
            for (int nt = 0; nt < 4; ++nt)
                acc[ms][nt] = __builtin_amdgcn_wmma_f32_16x16x32_bf16(
                    false, aF[cb][ms], false, bF[cb][nt], (short)0,
                    acc[ms][nt], false, false);
    }

    // ------------------------------ epilogues ------------------------------
    if constexpr (EPI == EPI_GELU || EPI == EPI_WARP) {
#pragma unroll
        for (int ms = 0; ms < 2; ++ms)
#pragma unroll
        for (int nt = 0; nt < 4; ++nt) {
            const int col  = n0 + ng * 64 + nt * 16 + l15;
            const float bc = bias[col];
#pragma unroll
            for (int r = 0; r < 8; ++r) {
                const int m = m0 + mg * 32 + ms * 16 + hi * 8 + r;
                const float v = acc[ms][nt][r] + bc;
                if constexpr (EPI == EPI_GELU) {
                    outBF[(size_t)m * NN + col] = (__bf16)geluf(v);
                } else {
                    const float xw = resid[(size_t)m * DMODEL + col] * v;
                    outBF[(size_t)m * DMODEL + col] = (__bf16)xw;
                }
            }
        }
    } else {
        // stage acc + bias + residual into LDS (full 256-wide rows -> local LN)
#pragma unroll
        for (int ms = 0; ms < 2; ++ms)
#pragma unroll
        for (int nt = 0; nt < 4; ++nt) {
            const int col  = ng * 64 + nt * 16 + l15;   // n0 == 0 for LN kernels
            const float bc = bias[col];
#pragma unroll
            for (int r = 0; r < 8; ++r) {
                const int m = mg * 32 + ms * 16 + hi * 8 + r;
                Cs[m * CS_ST + col] =
                    acc[ms][nt][r] + bc + resid[(size_t)(m0 + m) * DMODEL + col];
            }
        }
        __syncthreads();
        const int r = tid >> 2, q = tid & 3;   // 4 threads per row
        float s1 = 0.f, s2 = 0.f;
#pragma unroll 8
        for (int c = q * 64; c < q * 64 + 64; ++c) {
            const float v = Cs[r * CS_ST + c];
            s1 += v; s2 += v * v;
        }
        red[(r * 4 + q) * 2 + 0] = s1;
        red[(r * 4 + q) * 2 + 1] = s2;
        __syncthreads();
        float t1 = 0.f, t2 = 0.f;
#pragma unroll
        for (int j = 0; j < 4; ++j) {
            t1 += red[(r * 4 + j) * 2 + 0];
            t2 += red[(r * 4 + j) * 2 + 1];
        }
        const float mean = t1 * (1.0f / 256.0f);
        const float var  = t2 * (1.0f / 256.0f) - mean * mean;
        const float rstd = rsqrtf(var + 1e-5f);
#pragma unroll 8
        for (int c = q * 64; c < q * 64 + 64; ++c) {
            const float o = (Cs[r * CS_ST + c] - mean) * rstd * gamma[c] + beta[c];
            const size_t idx = (size_t)(m0 + r) * DMODEL + c;
            outF[idx] = o;
            if constexpr (EPI == EPI_MIX_LN) outBF[idx] = (__bf16)o;
        }
    }
}

// ---------------------------------------------------------------------------
// Host launcher
// ---------------------------------------------------------------------------
extern "C" void kernel_launch(void* const* d_in, const int* in_sizes, int n_in,
                              void* d_out, int out_size, void* d_ws, size_t ws_size,
                              hipStream_t stream) {
    (void)in_sizes; (void)n_in; (void)out_size; (void)ws_size;

    const float* x    = (const float*)d_in[0];
    const float* wg1  = (const float*)d_in[1];
    const float* bg1  = (const float*)d_in[2];
    const float* wg2  = (const float*)d_in[3];
    const float* bg2  = (const float*)d_in[4];
    const float* wmix = (const float*)d_in[5];
    const float* bmix = (const float*)d_in[6];
    const float* g1   = (const float*)d_in[7];
    const float* be1  = (const float*)d_in[8];
    const float* wf1  = (const float*)d_in[9];
    const float* bf1  = (const float*)d_in[10];
    const float* wf2  = (const float*)d_in[11];
    const float* bf2  = (const float*)d_in[12];
    const float* g2   = (const float*)d_in[13];
    const float* be2  = (const float*)d_in[14];
    float* out = (float*)d_out;

    // ---- workspace layout ----
    char* ws = (char*)d_ws;
    size_t off = 0;
    auto carve = [&](size_t bytes) { char* p = ws + off; off += (bytes + 255) & ~size_t(255); return p; };
    __bf16* xb    = (__bf16*)carve((size_t)TOK * DMODEL * 2);   // x bf16 (dead after k1)
    __bf16* h_bf  = (__bf16*)carve((size_t)TOK * DMODEL * 2);
    __bf16* xw_bf = (__bf16*)carve((size_t)TOK * DMODEL * 2);
    float*  y_f32 = (float*) carve((size_t)TOK * DMODEL * 4);
    __bf16* gl_bf = (__bf16*)carve((size_t)TOK * DFFN * 2);
    __bf16* wg1b  = (__bf16*)carve((size_t)DMODEL * DMODEL * 2);
    __bf16* wg2b  = (__bf16*)carve((size_t)DMODEL * DMODEL * 2);
    __bf16* wmixb = (__bf16*)carve((size_t)4 * DMODEL * DMODEL * 2); // first 1024 rows
    __bf16* wf1b  = (__bf16*)carve((size_t)DMODEL * DFFN * 2);
    __bf16* wf2b  = (__bf16*)carve((size_t)DFFN * DMODEL * 2);
    __bf16* y_bf  = xb;   // alias: xb dead before y_bf is produced

    const size_t SH_LN = (size_t)64 * CS_ST * 4 + (size_t)64 * 4 * 2 * 4;

    // ---- activation convert + weight shuffles ----
    cvt_bf16_k<<<(TOK * DMODEL + 255) / 256, 256, 0, stream>>>(x, xb, TOK * DMODEL);
    auto shuf = [&](const float* W, __bf16* o, int K, int NN) {
        shuffle_b_k<<<(K * NN / 2 + 255) / 256, 256, 0, stream>>>(W, o, K, NN);
    };
    shuf(wg1,  wg1b,  DMODEL, DMODEL);
    shuf(wg2,  wg2b,  DMODEL, DMODEL);
    shuf(wmix, wmixb, 4 * DMODEL, DMODEL);   // only the past-window rows matter
    shuf(wf1,  wf1b,  DMODEL, DFFN);
    shuf(wf2,  wf2b,  DFFN, DMODEL);

    const dim3 blk(256);
    const dim3 grid_d(TOK / 64, 1);
    const dim3 grid_f(TOK / 64, DFFN / 256);

    // 1) h = gelu(x @ wg1 + bg1)
    gemm64x256_k<EPI_GELU, false, DMODEL, DMODEL><<<grid_d, blk, 0, stream>>>(
        xb, wg1b, bg1, nullptr, nullptr, nullptr, nullptr, h_bf);
    // 2) xw = x * (h @ wg2 + bg2)
    gemm64x256_k<EPI_WARP, false, DMODEL, DMODEL><<<grid_d, blk, 0, stream>>>(
        h_bf, wg2b, bg2, x, nullptr, nullptr, nullptr, xw_bf);
    // 3) y = LN(x + local_mix(xw) @ wmix + bmix)   [shifted-window GEMM, K=1024]
    gemm64x256_k<EPI_MIX_LN, true, 4 * DMODEL, DMODEL><<<grid_d, blk, SH_LN, stream>>>(
        xw_bf, wmixb, bmix, x, g1, be1, y_f32, y_bf);
    // 4) g = gelu(y @ wf1 + bf1)
    gemm64x256_k<EPI_GELU, false, DMODEL, DFFN><<<grid_f, blk, 0, stream>>>(
        y_bf, wf1b, bf1, nullptr, nullptr, nullptr, nullptr, gl_bf);
    // 5) out = LN(y + g @ wf2 + bf2)
    gemm64x256_k<EPI_FFN_LN, false, DFFN, DMODEL><<<grid_d, blk, SH_LN, stream>>>(
        gl_bf, wf2b, bf2, y_f32, g2, be2, out, nullptr);
}
